// BitLinear_67190468378670
// MI455X (gfx1250) — matrix operand order repaired and checked by reading, not compiled, not run on hardware
//
#include <hip/hip_runtime.h>
#include <hip/hip_bf16.h>

typedef __attribute__((ext_vector_type(8)))  __bf16 bf16x8;
typedef __attribute__((ext_vector_type(16))) __bf16 bf16x16;
typedef __attribute__((ext_vector_type(8)))  float  f32x8;
typedef __attribute__((ext_vector_type(4)))  int    v4i;

static constexpr int M = 8192;    // 4 * 2048
static constexpr int K = 4096;
static constexpr int N = 16384;
static constexpr float TQ_EPS = 1e-8f;

static constexpr int BM = 128, BN = 128, BK = 64;
static constexpr int LDT = 72;    // LDS row pitch in halves (16B-aligned, padded vs 64)

#define AS1 __attribute__((address_space(1)))
#define AS3 __attribute__((address_space(3)))

// ---------- CDNA5 async global->LDS copy (ASYNCcnt-tracked, no VGPR bounce) ----------
__device__ __forceinline__ void async_copy_b128(const unsigned short* g, unsigned short* l) {
#if __has_builtin(__builtin_amdgcn_global_load_async_to_lds_b128)
    __builtin_amdgcn_global_load_async_to_lds_b128((AS1 v4i*)g, (AS3 v4i*)l, 0, 0);
#else
    unsigned lds_off = (unsigned)(unsigned long long)(AS3 void*)l;
    asm volatile("global_load_async_to_lds_b128 %0, %1, off"
                 :: "v"(lds_off), "v"((unsigned long long)g) : "memory");
#endif
}

template <int N_>
__device__ __forceinline__ void wait_asynccnt() {
#if __has_builtin(__builtin_amdgcn_s_wait_asynccnt)
    __builtin_amdgcn_s_wait_asynccnt(N_);
#else
    asm volatile("s_wait_asynccnt %0" :: "i"(N_) : "memory");
#endif
}

// ---------- helpers ----------
__device__ inline unsigned short f32_to_bf16_rne(float f) {
    unsigned u = __float_as_uint(f);
    return (unsigned short)((u + 0x7FFFu + ((u >> 16) & 1u)) >> 16);
}
// exact for {-1, 0, +1}: truncate mantissa (all-zero low bits)
__device__ inline unsigned short ternary_bf16(float v, float inv) {
    float q = fminf(1.f, fmaxf(-1.f, rintf(v * inv)));   // rintf == RNE, matches jnp.round
    return (unsigned short)(__float_as_uint(q) >> 16);
}
__device__ inline bf16x16 make_frag(const __bf16* lo, const __bf16* hi) {
    bf16x8 a = *(const bf16x8*)lo;
    bf16x8 b = *(const bf16x8*)hi;
    return __builtin_shufflevector(a, b, 0,1,2,3,4,5,6,7,8,9,10,11,12,13,14,15);
}

// ---------- pass 1: deterministic |W| partial sums ----------
__global__ __launch_bounds__(256) void absmean_part(const float4* __restrict__ w,
                                                    float* __restrict__ partials) {
    __shared__ float sdata[256];
    const int tid = threadIdx.x;
    const size_t nvec   = (size_t)N * K / 4;            // 16,777,216
    const size_t stride = (size_t)gridDim.x * 256;
    float s = 0.f;
    for (size_t i = (size_t)blockIdx.x * 256 + tid; i < nvec; i += stride) {
        float4 v = w[i];
        s += fabsf(v.x) + fabsf(v.y) + fabsf(v.z) + fabsf(v.w);
    }
    sdata[tid] = s;
    __syncthreads();
    for (int off = 128; off > 0; off >>= 1) {
        if (tid < off) sdata[tid] += sdata[tid + off];
        __syncthreads();
    }
    if (tid == 0) partials[blockIdx.x] = sdata[0];
}

// ---------- pass 2: final reduce -> alpha ----------
__global__ __launch_bounds__(1024) void absmean_final(const float* __restrict__ partials,
                                                      float* __restrict__ alpha) {
    __shared__ float sdata[1024];
    const int tid = threadIdx.x;
    float s = 0.f;
    for (int i = tid; i < 4096; i += 1024) s += partials[i];
    sdata[tid] = s;
    __syncthreads();
    for (int off = 512; off > 0; off >>= 1) {
        if (tid < off) sdata[tid] += sdata[tid + off];
        __syncthreads();
    }
    if (tid == 0) *alpha = sdata[0] / (float)((size_t)N * K);
}

// ---------- quantize W -> ternary bf16 ----------
__global__ __launch_bounds__(256) void quantize_w(const float4* __restrict__ w,
                                                  ushort4* __restrict__ wq,
                                                  const float* __restrict__ alphap) {
    const float inv = 1.f / (*alphap + TQ_EPS);
    const size_t gid = (size_t)blockIdx.x * 256 + threadIdx.x;
    float4 v = w[gid];
    ushort4 o;
    o.x = ternary_bf16(v.x, inv);
    o.y = ternary_bf16(v.y, inv);
    o.z = ternary_bf16(v.z, inv);
    o.w = ternary_bf16(v.w, inv);
    wq[gid] = o;
}

// ---------- convert x -> bf16 ----------
__global__ __launch_bounds__(256) void convert_x(const float4* __restrict__ x,
                                                 ushort4* __restrict__ xb) {
    const size_t gid = (size_t)blockIdx.x * 256 + threadIdx.x;
    float4 v = x[gid];
    ushort4 o;
    o.x = f32_to_bf16_rne(v.x);
    o.y = f32_to_bf16_rne(v.y);
    o.z = f32_to_bf16_rne(v.z);
    o.w = f32_to_bf16_rne(v.w);
    xb[gid] = o;
}

// ---------- WMMA GEMM: out = alpha * (xb @ wq^T) ----------
// Block tile 128x128, BK=64, double-buffered LDS filled by async global->LDS loads.
// 8 waves in 2(M) x 4(N); each wave computes 4x2 tiles of 16x16 via v_wmma_f32_16x16x32_bf16.
__global__ __launch_bounds__(256) void gemm_ternary(const unsigned short* __restrict__ xb,
                                                    const unsigned short* __restrict__ wq,
                                                    float* __restrict__ out,
                                                    const float* __restrict__ alphap) {
    __shared__ __align__(16) unsigned short As[2 * BM * LDT];
    __shared__ __align__(16) unsigned short Bs[2 * BN * LDT];

    const int tid   = threadIdx.x;
    const int wave  = tid >> 5;
    const int lane  = tid & 31;
    const int wm    = wave >> 2;          // 0..1  -> 64-row slab
    const int wn    = wave & 3;           // 0..3  -> 32-col slab
    const int mlane = lane & 15;
    const int khalf = lane >> 4;          // 0/1 lane-half per ISA 16-bit layouts

    const int bm = blockIdx.y * BM;
    const int bn = blockIdx.x * BN;

    f32x8 acc[4][2] = {};

    // issue one tile: 128 rows x 64 halves x 2 mats = 2048 x 16B chunks; 8 async loads/thread
    auto issue_tile = [&](int buf, int kt) {
        unsigned short* Ad = As + buf * (BM * LDT);
        unsigned short* Bd = Bs + buf * (BM * LDT);
        #pragma unroll
        for (int i = 0; i < 4; ++i) {
            const int c   = i * 256 + tid;
            const int row = c >> 3;
            const int chk = c & 7;
            async_copy_b128(xb + (size_t)(bm + row) * K + kt + chk * 8,
                            Ad + row * LDT + chk * 8);
            async_copy_b128(wq + (size_t)(bn + row) * K + kt + chk * 8,
                            Bd + row * LDT + chk * 8);
        }
    };

    constexpr int NITER = K / BK;         // 64
    issue_tile(0, 0);

    for (int it = 0; it < NITER; ++it) {
        const int cur = it & 1;
        if (it + 1 < NITER) {
            issue_tile(1 - cur, (it + 1) * BK);   // prefetch next tile into other buffer
            wait_asynccnt<8>();                   // current tile's 8 loads complete
        } else {
            wait_asynccnt<0>();
        }
        __syncthreads();                          // all waves' current-tile loads done

        const unsigned short* Ab = As + cur * (BM * LDT);
        const unsigned short* Bb = Bs + cur * (BM * LDT);

        #pragma unroll
        for (int ks = 0; ks < BK; ks += 32) {
            bf16x16 af[4], bfr[2];
            // A 16x32 bf16: lane(m,khalf): VGPR0-3 = K[khalf*8..+8), VGPR4-7 = K[16+khalf*8..+8)
            #pragma unroll
            for (int i = 0; i < 4; ++i) {
                const __bf16* base =
                    (const __bf16*)&Ab[(wm * 64 + i * 16 + mlane) * LDT + ks + khalf * 8];
                af[i] = make_frag(base, base + 16);
            }
            // B 32x16 bf16: lane(n,khalf): VGPR0-3 = K[khalf*8..+8), VGPR4-7 = K[16+khalf*8..+8)
            #pragma unroll
            for (int j = 0; j < 2; ++j) {
                const __bf16* base =
                    (const __bf16*)&Bb[(wn * 32 + j * 16 + mlane) * LDT + ks + khalf * 8];
                bfr[j] = make_frag(base, base + 16);
            }
            #pragma unroll
            for (int i = 0; i < 4; ++i)
                #pragma unroll
                for (int j = 0; j < 2; ++j)
                    acc[i][j] = __builtin_amdgcn_wmma_f32_16x16x32_bf16(
                        false, af[i], false, bfr[j], (short)0, acc[i][j], false, false);
        }
        __syncthreads();                          // protect buffer before next-iter overwrite
    }

    const float alpha = *alphap;
    // C/D layout: VGPR r, lanes 0-15 -> M = r, lanes 16-31 -> M = r+8; N = lane&15
    #pragma unroll
    for (int i = 0; i < 4; ++i) {
        const int row0 = bm + wm * 64 + i * 16 + khalf * 8;
        #pragma unroll
        for (int j = 0; j < 2; ++j) {
            const int col = bn + wn * 32 + j * 16 + mlane;
            #pragma unroll
            for (int r = 0; r < 8; ++r)
                out[(size_t)(row0 + r) * N + col] = alpha * acc[i][j][r];
        }
    }
}

extern "C" void kernel_launch(void* const* d_in, const int* in_sizes, int n_in,
                              void* d_out, int out_size, void* d_ws, size_t ws_size,
                              hipStream_t stream) {
    const float* x = (const float*)d_in[0];   // [4,2048,4096] f32
    const float* w = (const float*)d_in[1];   // [16384,4096]  f32
    float* out = (float*)d_out;               // [4,2048,16384] f32

    // workspace layout: alpha | partials[4096] | wq bf16 (128MiB) | xb bf16 (64MiB)
    char* ws = (char*)d_ws;
    float* alpha    = (float*)ws;
    float* partials = (float*)(ws + 256);
    unsigned short* wq = (unsigned short*)(ws + 32768);
    unsigned short* xb = wq + (size_t)N * K;

    absmean_part<<<4096, 256, 0, stream>>>((const float4*)w, partials);
    absmean_final<<<1, 1024, 0, stream>>>(partials, alpha);
    quantize_w<<<(int)(((size_t)N * K / 4) / 256), 256, 0, stream>>>(
        (const float4*)w, (ushort4*)wq, alpha);
    convert_x<<<(int)(((size_t)M * K / 4) / 256), 256, 0, stream>>>(
        (const float4*)x, (ushort4*)xb);

    dim3 grid(N / BN, M / BM);   // 128 x 64
    gemm_ternary<<<grid, 256, 0, stream>>>(xb, wq, out, alpha);
}
